// SpikingClassifierV2_61160334295367
// MI455X (gfx1250) — matrix operand order repaired and checked by reading, not compiled, not run on hardware
//
#include <hip/hip_runtime.h>
#include <hip/hip_bf16.h>

// ---------------------------------------------------------------------------
// SpikingClassifierV2 forward for gfx1250 (MI455X).
// bf16 WMMA (v_wmma_f32_16x16x32_bf16) for conv2 implicit GEMM, channel_proj,
// fc1 and fc2. conv2 uses double-buffered async-to-LDS staging (ASYNCcnt) when
// the toolchain exposes the gfx1250 builtins, else VGPR staging fallback.
// ---------------------------------------------------------------------------

typedef __attribute__((ext_vector_type(16))) __bf16 v16bf;
typedef __attribute__((ext_vector_type(8)))  float  v8f;

union BF16Frag {
  v16bf v;
  unsigned short s[16];
  uint4 u[2];
};

#define EPSF 1e-5f

#if defined(__AMDGCN__) && \
    __has_builtin(__builtin_amdgcn_global_load_async_to_lds_b128) && \
    __has_builtin(__builtin_amdgcn_s_wait_asynccnt)
#define USE_ASYNC_LDS 1
#define GLOBAL_AS __attribute__((address_space(1)))
#define LDS_AS    __attribute__((address_space(3)))
typedef int v4i __attribute__((vector_size(16)));   // builtin's b128 payload type
#endif
#ifndef USE_ASYNC_LDS
#define USE_ASYNC_LDS 0
#endif

__device__ __forceinline__ unsigned short f2bf(float f) {
  unsigned u = __float_as_uint(f);
  return (unsigned short)((u + 0x7FFFu + ((u >> 16) & 1u)) >> 16);
}

// ---------------- workspace layout (bytes, all 256-aligned) ----------------
static const size_t OFF_H1   = 0;                        // 64*64*200*32 bf16
static const size_t OFF_WP   = OFF_H1   + 52428800;      // conv2 wpack
static const size_t OFF_PW   = OFF_WP   + 36864;         // pw pack
static const size_t OFF_F1P  = OFF_PW   + 32768;         // fc1 pack
static const size_t OFF_F2P  = OFF_F1P  + 131072;        // fc2 pack
static const size_t OFF_HM   = OFF_F2P  + 65536;         // hmean (B,100,64) f32
static const size_t OFF_PRAW = OFF_HM   + 1638400;       // proj raw (B,100,256)
static const size_t OFF_PROJ = OFF_PRAW + 6553600;       // proj post-LN
static const size_t OFF_CUR  = OFF_PROJ + 6553600;       // cur (B,50,256)
static const size_t OFF_CUR1 = OFF_CUR  + 3276800;       // cur1 (B,50,256)
static const size_t OFF_S1B  = OFF_CUR1 + 3276800;       // s1 spikes bf16
static const size_t OFF_CUR2 = OFF_S1B  + 1638400;       // cur2 (B,50,128)
static const size_t OFF_S2   = OFF_CUR2 + 1638400;       // s2 (B,50,128)
static const size_t OFF_CUR3 = OFF_S2   + 1638400;       // cur3 (B,50,2)

// ---------------------------------------------------------------------------
// conv1 (1->32, 3x3 SAME) + BN(eval) + ReLU + maxpool2 -> h1 bf16 NHWC
// out: (B,64,200,32)
// ---------------------------------------------------------------------------
__global__ void k_conv1(const float* __restrict__ x,  const float* __restrict__ w,
                        const float* __restrict__ cb, const float* __restrict__ g,
                        const float* __restrict__ bb, const float* __restrict__ m,
                        const float* __restrict__ vv, unsigned short* __restrict__ h1) {
  int idx = blockIdx.x * blockDim.x + threadIdx.x;
  if (idx >= 64 * 64 * 200 * 32) return;
  int c    = idx & 31;
  int rest = idx >> 5;
  int wo   = rest % 200;
  int t2   = rest / 200;
  int ho   = t2 & 63;
  int b    = t2 >> 6;

  float wk[9];
#pragma unroll
  for (int t = 0; t < 9; ++t) wk[t] = w[c * 9 + t];
  float scale = g[c] * rsqrtf(vv[c] + EPSF);
  float shift = bb[c] + (cb[c] - m[c]) * scale;

  float win[4][4];
#pragma unroll
  for (int i = 0; i < 4; ++i) {
    int yy = 2 * ho - 1 + i;
#pragma unroll
    for (int j = 0; j < 4; ++j) {
      int xx = 2 * wo - 1 + j;
      win[i][j] = (yy >= 0 && yy < 128 && xx >= 0 && xx < 400)
                      ? x[((size_t)b * 128 + yy) * 400 + xx] : 0.f;
    }
  }
  float best = 0.f;
#pragma unroll
  for (int dy = 0; dy < 2; ++dy)
#pragma unroll
    for (int dx = 0; dx < 2; ++dx) {
      float s = 0.f;
#pragma unroll
      for (int ky = 0; ky < 3; ++ky)
#pragma unroll
        for (int kx = 0; kx < 3; ++kx)
          s += win[dy + ky][dx + kx] * wk[ky * 3 + kx];
      best = fmaxf(best, fmaxf(s * scale + shift, 0.f));
    }
  h1[(((size_t)b * 64 + ho) * 200 + wo) * 32 + c] = f2bf(best);
}

// ---------------------------------------------------------------------------
// conv2 weight repack: (O=64,I=32,3,3) f32 -> wpack[tap][ntile][lane][16] bf16
// ---------------------------------------------------------------------------
__global__ void k_packw_conv(const float* __restrict__ w, unsigned short* __restrict__ wp) {
  int idx = blockIdx.x * blockDim.x + threadIdx.x;
  if (idx >= 9 * 4 * 32 * 16) return;
  int j    = idx & 15;
  int lane = (idx >> 4) & 31;
  int nt   = (idx >> 9) & 3;
  int tap  = idx >> 11;
  int k    = (j < 8) ? ((lane >> 4) * 8 + j) : (16 + (lane >> 4) * 8 + (j - 8));  // input channel
  int n    = nt * 16 + (lane & 15);               // output channel
  int ky = tap / 3, kx = tap % 3;
  wp[idx] = f2bf(w[((n * 32 + k) * 3 + ky) * 3 + kx]);
}

// generic GEMM weight repack: src (K,N) row-major f32 -> [ks][ntile][lane][16] bf16
__global__ void k_packw_gemm(const float* __restrict__ src, unsigned short* __restrict__ dst,
                             int N, int total) {
  int idx = blockIdx.x * blockDim.x + threadIdx.x;
  if (idx >= total) return;
  int j    = idx & 15;
  int lane = (idx >> 4) & 31;
  int rest = idx >> 9;          // ks*NT + nt
  int NT   = N >> 4;
  int nt   = rest % NT;
  int ks   = rest / NT;
  int k = ks * 32 + ((j < 8) ? ((lane >> 4) * 8 + j) : (16 + (lane >> 4) * 8 + (j - 8)));
  int n = nt * 16 + (lane & 15);
  dst[idx] = f2bf(src[(size_t)k * N + n]);
}

// ---------------------------------------------------------------------------
// conv2 activation staging: 4 rows x 18 cols x 32ch bf16 -> LDS buffer.
// Async-to-LDS (ASYNCcnt) when available, VGPR staging otherwise.
// ---------------------------------------------------------------------------
__device__ __forceinline__ void stage_rows(const unsigned short* __restrict__ h1,
                                           unsigned short* sAbuf, int tid,
                                           int b, int xt, int y0) {
  for (int s = tid; s < 72; s += 128) {
    int rr = s / 18, cc = s % 18;
    int ry = y0 - 1 + rr;
    int cx = xt * 16 - 1 + cc;
    unsigned short* dp = &sAbuf[(rr * 18 + cc) * 32];
    if (ry >= 0 && ry < 64 && cx >= 0 && cx < 200) {
      const unsigned short* sp = &h1[(((size_t)b * 64 + ry) * 200 + cx) * 32];
#if USE_ASYNC_LDS
#pragma unroll
      for (int q = 0; q < 4; ++q)
        __builtin_amdgcn_global_load_async_to_lds_b128(
            (GLOBAL_AS v4i*)(sp + q * 8), (LDS_AS v4i*)(dp + q * 8), 0, 0);
#else
#pragma unroll
      for (int q = 0; q < 4; ++q)
        ((uint4*)dp)[q] = ((const uint4*)sp)[q];
#endif
    } else {
      uint4 z = {0u, 0u, 0u, 0u};
#pragma unroll
      for (int q = 0; q < 4; ++q) ((uint4*)dp)[q] = z;
    }
  }
}

// ---------------------------------------------------------------------------
// conv2 implicit GEMM (WMMA bf16) + BN + ReLU + maxpool2 + freq-mean
// grid: 64 b * 13 xtiles, block = 128 (4 waves = 4 output-channel tiles)
// Double-buffered LDS pipeline: stage f+1 while computing f.
// output: hmean (B, T=100, C=64) f32
// ---------------------------------------------------------------------------
__global__ void __launch_bounds__(128) k_conv2(
    const unsigned short* __restrict__ h1, const unsigned short* __restrict__ wp,
    const float* __restrict__ cb, const float* __restrict__ g,
    const float* __restrict__ bb, const float* __restrict__ m,
    const float* __restrict__ vv, float* __restrict__ hmean) {
  __shared__ __align__(16) unsigned short sW[9 * 4 * 32 * 16];   // 36864 B
  __shared__ __align__(16) unsigned short sA[2 * 4 * 18 * 32];   // 2 x 4608 B

  int tid  = threadIdx.x;
  int b    = blockIdx.x / 13;
  int xt   = blockIdx.x % 13;
  int lane = tid & 31;
  int wv   = tid >> 5;          // ntile (output-channel tile)
  int p    = lane & 15;
  int half = lane >> 4;

  // stage packed weights into LDS once
  for (int i = tid; i < (9 * 4 * 32 * 16) / 8; i += 128) {
#if USE_ASYNC_LDS
    __builtin_amdgcn_global_load_async_to_lds_b128(
        (GLOBAL_AS v4i*)((const uint4*)wp + i), (LDS_AS v4i*)((uint4*)sW + i), 0, 0);
#else
    ((uint4*)sW)[i] = ((const uint4*)wp)[i];
#endif
  }

  int n = wv * 16 + p;
  float scale = g[n] * rsqrtf(vv[n] + EPSF);
  float shift = bb[n] + (cb[n] - m[n]) * scale;

  float macc[4] = {0.f, 0.f, 0.f, 0.f};

  // prologue: stage f = 0 into buffer 0
  stage_rows(h1, sA, tid, b, xt, 0);
#if USE_ASYNC_LDS
  __builtin_amdgcn_s_wait_asynccnt(0);
#endif
  __syncthreads();

  for (int f = 0; f < 32; ++f) {
    unsigned short* curA = sA + (f & 1) * (4 * 18 * 32);
    // issue staging for next iteration into the other buffer (overlaps WMMA)
    if (f < 31)
      stage_rows(h1, sA + ((f + 1) & 1) * (4 * 18 * 32), tid, b, xt, 2 * (f + 1));

    float rowv[2][8];
#pragma unroll
    for (int r = 0; r < 2; ++r) {        // two conv output rows (for 2x2 pool)
      v8f acc = {};
#pragma unroll
      for (int tap = 0; tap < 9; ++tap) {
        int ky = tap / 3, kx = tap % 3;
        BF16Frag a, bm;
        int base = ((r + ky) * 18 + (p + kx)) * 32 + half * 8;
        a.u[0] = *(const uint4*)&curA[base];       // K {0..7}  (+8 upper half)
        a.u[1] = *(const uint4*)&curA[base + 16];  // K {16..23}(+8 upper half)
        int wb = ((tap * 4 + wv) * 32 + lane) * 16;
        bm.u[0] = *(const uint4*)&sW[wb];
        bm.u[1] = *(const uint4*)&sW[wb + 8];
        acc = __builtin_amdgcn_wmma_f32_16x16x32_bf16(
            false, a.v, false, bm.v, (short)0, acc, false, false);
      }
#pragma unroll
      for (int j = 0; j < 8; ++j)
        rowv[r][j] = fmaxf(acc[j] * scale + shift, 0.f);   // BN + ReLU
    }
    // 2x2 max-pool (x pairs within lane, y pair across rowv) + freq-mean accum
#pragma unroll
    for (int j2 = 0; j2 < 4; ++j2) {
      float pl = fmaxf(fmaxf(rowv[0][2 * j2], rowv[0][2 * j2 + 1]),
                       fmaxf(rowv[1][2 * j2], rowv[1][2 * j2 + 1]));
      macc[j2] += pl * (1.f / 32.f);
    }
#if USE_ASYNC_LDS
    __builtin_amdgcn_s_wait_asynccnt(0);
#endif
    __syncthreads();
  }
  // store mean: unique (b,t,c) per lane slot -> no atomics
#pragma unroll
  for (int j2 = 0; j2 < 4; ++j2) {
    int t = xt * 8 + half * 4 + j2;
    if (t < 100) hmean[((size_t)b * 100 + t) * 64 + n] = macc[j2];
  }
}

// ---------------------------------------------------------------------------
// generic WMMA GEMM: C(MxN) = A(MxK) * Bpack + bias.  A is f32 (converted) or
// bf16 (Ab != null). One wave per 16x16 tile.
// ---------------------------------------------------------------------------
__global__ void __launch_bounds__(256) k_gemm(
    const float* __restrict__ Af, const unsigned short* __restrict__ Ab,
    const unsigned short* __restrict__ Bp, const float* __restrict__ bias,
    float* __restrict__ C, int mtiles, int ntiles, int kSteps, int K, int N) {
  int waveId = blockIdx.x * (blockDim.x >> 5) + (threadIdx.x >> 5);
  if (waveId >= mtiles * ntiles) return;
  int mt = waveId / ntiles, nt = waveId % ntiles;
  int lane = threadIdx.x & 31;
  int p = lane & 15, half = lane >> 4;
  int mrow = mt * 16 + p;

  v8f acc = {};
  for (int ks = 0; ks < kSteps; ++ks) {
    BF16Frag a, bm;
    int k0 = ks * 32 + half * 8;
    if (Ab) {
      const unsigned short* rp = Ab + (size_t)mrow * K + k0;
      __builtin_prefetch(rp + 32, 0, 2);
      a.u[0] = *(const uint4*)rp;
      a.u[1] = *(const uint4*)(rp + 16);
    } else {
      const float* rp = Af + (size_t)mrow * K + k0;
      __builtin_prefetch(rp + 32, 0, 2);
      float4 f0 = *(const float4*)rp;
      float4 f1 = *(const float4*)(rp + 4);
      float4 f2 = *(const float4*)(rp + 16);
      float4 f3 = *(const float4*)(rp + 20);
      a.s[0] = f2bf(f0.x); a.s[1] = f2bf(f0.y); a.s[2]  = f2bf(f0.z); a.s[3]  = f2bf(f0.w);
      a.s[4] = f2bf(f1.x); a.s[5] = f2bf(f1.y); a.s[6]  = f2bf(f1.z); a.s[7]  = f2bf(f1.w);
      a.s[8] = f2bf(f2.x); a.s[9] = f2bf(f2.y); a.s[10] = f2bf(f2.z); a.s[11] = f2bf(f2.w);
      a.s[12] = f2bf(f3.x); a.s[13] = f2bf(f3.y); a.s[14] = f2bf(f3.z); a.s[15] = f2bf(f3.w);
    }
    const unsigned short* wb = Bp + (((size_t)ks * ntiles + nt) * 32 + lane) * 16;
    bm.u[0] = *(const uint4*)wb;
    bm.u[1] = *(const uint4*)(wb + 8);
    acc = __builtin_amdgcn_wmma_f32_16x16x32_bf16(
        false, a.v, false, bm.v, (short)0, acc, false, false);
  }
  int nc = nt * 16 + p;
  float bv = bias ? bias[nc] : 0.f;
#pragma unroll
  for (int j = 0; j < 8; ++j) {
    int row = mt * 16 + half * 8 + j;
    C[(size_t)row * N + nc] = acc[j] + bv;
  }
}

// ---------------------------------------------------------------------------
// LayerNorm over C=256, one wave per row
// ---------------------------------------------------------------------------
__global__ void __launch_bounds__(256) k_ln256(
    const float* __restrict__ in, float* __restrict__ out,
    const float* __restrict__ g, const float* __restrict__ bv, int rows) {
  int waveId = blockIdx.x * 8 + (threadIdx.x >> 5);
  if (waveId >= rows) return;
  int lane = threadIdx.x & 31;
  const float* rp = in + (size_t)waveId * 256 + lane * 8;
  float4 v0 = *(const float4*)rp;
  float4 v1 = *(const float4*)(rp + 4);
  float vals[8] = {v0.x, v0.y, v0.z, v0.w, v1.x, v1.y, v1.z, v1.w};
  float s = 0.f, sq = 0.f;
#pragma unroll
  for (int j = 0; j < 8; ++j) { s += vals[j]; sq += vals[j] * vals[j]; }
  for (int off = 16; off > 0; off >>= 1) {
    s  += __shfl_xor(s, off, 32);
    sq += __shfl_xor(sq, off, 32);
  }
  float mean = s * (1.f / 256.f);
  float var  = sq * (1.f / 256.f) - mean * mean;
  float inv  = rsqrtf(var + EPSF);
  float* op = out + (size_t)waveId * 256 + lane * 8;
#pragma unroll
  for (int j = 0; j < 8; ++j)
    op[j] = (vals[j] - mean) * inv * g[lane * 8 + j] + bv[lane * 8 + j];
}

// ---------------------------------------------------------------------------
// interp (Tin=100 -> 50 is exact pair-average) + LayerNorm(ing, inb) -> cur
// ---------------------------------------------------------------------------
__global__ void __launch_bounds__(256) k_interp_ln(
    const float* __restrict__ proj, const float* __restrict__ g,
    const float* __restrict__ bv, float* __restrict__ cur) {
  int waveId = blockIdx.x * 8 + (threadIdx.x >> 5);
  if (waveId >= 3200) return;
  int lane = threadIdx.x & 31;
  int b = waveId / 50, st = waveId % 50;
  const float* r0 = proj + ((size_t)b * 100 + 2 * st) * 256 + lane * 8;
  const float* r1 = r0 + 256;
  float vals[8];
#pragma unroll
  for (int j = 0; j < 8; ++j) vals[j] = 0.5f * (r0[j] + r1[j]);
  float s = 0.f, sq = 0.f;
#pragma unroll
  for (int j = 0; j < 8; ++j) { s += vals[j]; sq += vals[j] * vals[j]; }
  for (int off = 16; off > 0; off >>= 1) {
    s  += __shfl_xor(s, off, 32);
    sq += __shfl_xor(sq, off, 32);
  }
  float mean = s * (1.f / 256.f);
  float var  = sq * (1.f / 256.f) - mean * mean;
  float inv  = rsqrtf(var + EPSF);
  float* op = cur + (size_t)waveId * 256 + lane * 8;
#pragma unroll
  for (int j = 0; j < 8; ++j)
    op[j] = (vals[j] - mean) * inv * g[lane * 8 + j] + bv[lane * 8 + j];
}

// ---------------------------------------------------------------------------
// LIF layer 1: elementwise scan over t; spikes -> d_out (f32) and bf16 buffer
// ---------------------------------------------------------------------------
__global__ void k_lif1(const float* __restrict__ cur1, const float* __restrict__ beta,
                       const float* __restrict__ thr, float* __restrict__ spk_out,
                       unsigned short* __restrict__ s1b) {
  int idx = blockIdx.x * blockDim.x + threadIdx.x;
  if (idx >= 64 * 256) return;
  int h = idx & 255, b = idx >> 8;
  float bt = fminf(fmaxf(beta[0], 0.f), 1.f);
  float th = thr[0];
  float mem = 0.f;
  for (int t = 0; t < 50; ++t) {
    size_t o = ((size_t)b * 50 + t) * 256 + h;
    float c = cur1[o];
    float reset = (mem > th) ? 1.f : 0.f;
    mem = bt * mem + c - reset * th;
    float spk = (mem - th > 0.f) ? 1.f : 0.f;
    spk_out[o] = spk;
    s1b[o] = (spk > 0.f) ? (unsigned short)0x3F80 : (unsigned short)0;  // bf16 1/0
  }
}

// LIF layer 2: (B,128) elementwise scan -> s2
__global__ void k_lif2(const float* __restrict__ cur2, const float* __restrict__ beta,
                       const float* __restrict__ thr, float* __restrict__ s2) {
  int idx = blockIdx.x * blockDim.x + threadIdx.x;
  if (idx >= 64 * 128) return;
  int h = idx & 127, b = idx >> 7;
  float bt = fminf(fmaxf(beta[0], 0.f), 1.f);
  float th = thr[0];
  float mem = 0.f;
  for (int t = 0; t < 50; ++t) {
    size_t o = ((size_t)b * 50 + t) * 128 + h;
    float c = cur2[o];
    float reset = (mem > th) ? 1.f : 0.f;
    mem = bt * mem + c - reset * th;
    s2[o] = (mem - th > 0.f) ? 1.f : 0.f;
  }
}

// fco: (B*50, 128) @ (128, 2) + bias -> cur3
__global__ void k_fco(const float* __restrict__ s2, const float* __restrict__ w,
                      const float* __restrict__ bias, float* __restrict__ cur3) {
  int idx = blockIdx.x * blockDim.x + threadIdx.x;
  if (idx >= 64 * 50 * 2) return;
  int cls = idx & 1;
  int bt  = idx >> 1;
  float s = bias[cls];
  const float* rp = s2 + (size_t)bt * 128;
  for (int k = 0; k < 128; ++k) s += rp[k] * w[k * 2 + cls];
  cur3[idx] = s;
}

// LIF layer 3: (B,2) scan, logits = sum_t mem_t
__global__ void k_lif3(const float* __restrict__ cur3, const float* __restrict__ beta,
                       const float* __restrict__ thr, float* __restrict__ logits) {
  int idx = blockIdx.x * blockDim.x + threadIdx.x;
  if (idx >= 128) return;
  int cls = idx & 1, b = idx >> 1;
  float bt = fminf(fmaxf(beta[0], 0.f), 1.f);
  float th = thr[0];
  float mem = 0.f, acc = 0.f;
  for (int t = 0; t < 50; ++t) {
    float c = cur3[((size_t)b * 50 + t) * 2 + cls];
    float reset = (mem > th) ? 1.f : 0.f;
    mem = bt * mem + c - reset * th;
    acc += mem;
  }
  logits[idx] = acc;
}

// ---------------------------------------------------------------------------
extern "C" void kernel_launch(void* const* d_in, const int* in_sizes, int n_in,
                              void* d_out, int out_size, void* d_ws, size_t ws_size,
                              hipStream_t stream) {
  (void)in_sizes; (void)n_in; (void)out_size; (void)ws_size;
  const float* x     = (const float*)d_in[0];
  const float* c1w   = (const float*)d_in[1];
  const float* c1b   = (const float*)d_in[2];
  const float* bn1g  = (const float*)d_in[3];
  const float* bn1b  = (const float*)d_in[4];
  const float* bn1m  = (const float*)d_in[5];
  const float* bn1v  = (const float*)d_in[6];
  const float* c2w   = (const float*)d_in[7];
  const float* c2b   = (const float*)d_in[8];
  const float* bn2g  = (const float*)d_in[9];
  const float* bn2b  = (const float*)d_in[10];
  const float* bn2m  = (const float*)d_in[11];
  const float* bn2v  = (const float*)d_in[12];
  const float* pw    = (const float*)d_in[13];
  const float* pb    = (const float*)d_in[14];
  const float* plg   = (const float*)d_in[15];
  const float* plb   = (const float*)d_in[16];
  const float* ing   = (const float*)d_in[17];
  const float* inb   = (const float*)d_in[18];
  const float* fc1w  = (const float*)d_in[19];
  const float* fc1b  = (const float*)d_in[20];
  const float* fc2w  = (const float*)d_in[21];
  const float* fc2b  = (const float*)d_in[22];
  const float* fcow  = (const float*)d_in[23];
  const float* fcob  = (const float*)d_in[24];
  const float* beta1 = (const float*)d_in[25];
  const float* thr1  = (const float*)d_in[26];
  const float* beta2 = (const float*)d_in[27];
  const float* thr2  = (const float*)d_in[28];
  const float* beta3 = (const float*)d_in[29];
  const float* thr3  = (const float*)d_in[30];

  char* ws = (char*)d_ws;
  unsigned short* h1    = (unsigned short*)(ws + OFF_H1);
  unsigned short* wpack = (unsigned short*)(ws + OFF_WP);
  unsigned short* pwp   = (unsigned short*)(ws + OFF_PW);
  unsigned short* f1p   = (unsigned short*)(ws + OFF_F1P);
  unsigned short* f2p   = (unsigned short*)(ws + OFF_F2P);
  float* hmean   = (float*)(ws + OFF_HM);
  float* projraw = (float*)(ws + OFF_PRAW);
  float* proj    = (float*)(ws + OFF_PROJ);
  float* cur     = (float*)(ws + OFF_CUR);
  float* cur1    = (float*)(ws + OFF_CUR1);
  unsigned short* s1b = (unsigned short*)(ws + OFF_S1B);
  float* cur2    = (float*)(ws + OFF_CUR2);
  float* s2      = (float*)(ws + OFF_S2);
  float* cur3    = (float*)(ws + OFF_CUR3);

  float* out    = (float*)d_out;
  float* logits = out;          // (64,2)
  float* spikes = out + 128;    // (64,50,256)

  // encoder stage 1 (VALU conv, K=9)
  k_conv1<<<(64 * 64 * 200 * 32 + 255) / 256, 256, 0, stream>>>(
      x, c1w, c1b, bn1g, bn1b, bn1m, bn1v, h1);

  // weight repacks for WMMA B-operands
  k_packw_conv<<<(9 * 4 * 32 * 16 + 255) / 256, 256, 0, stream>>>(c2w, wpack);
  k_packw_gemm<<<(16384 + 255) / 256, 256, 0, stream>>>(pw,   pwp, 256, 16384);
  k_packw_gemm<<<(65536 + 255) / 256, 256, 0, stream>>>(fc1w, f1p, 256, 65536);
  k_packw_gemm<<<(32768 + 255) / 256, 256, 0, stream>>>(fc2w, f2p, 128, 32768);

  // conv2 implicit GEMM (WMMA, async double-buffered LDS) + BN+ReLU+pool+mean
  k_conv2<<<64 * 13, 128, 0, stream>>>(h1, wpack, c2b, bn2g, bn2b, bn2m, bn2v, hmean);

  // channel_proj: (6400,64)@(64,256) + pb  (WMMA)
  k_gemm<<<800, 256, 0, stream>>>(hmean, nullptr, pwp, pb, projraw, 400, 16, 2, 64, 256);
  k_ln256<<<800, 256, 0, stream>>>(projraw, proj, plg, plb, 6400);

  // interp (pair average) + input LN
  k_interp_ln<<<400, 256, 0, stream>>>(proj, ing, inb, cur);

  // fc1: (3200,256)@(256,256) + fc1b  (WMMA)
  k_gemm<<<400, 256, 0, stream>>>(cur, nullptr, f1p, fc1b, cur1, 200, 16, 8, 256, 256);

  // layer-1 LIF scan -> spike_record (d_out) + bf16 spikes
  k_lif1<<<(64 * 256 + 255) / 256, 256, 0, stream>>>(cur1, beta1, thr1, spikes, s1b);

  // fc2 over all steps: (3200,256)@(256,128) + fc2b  (WMMA, bf16 A)
  k_gemm<<<200, 256, 0, stream>>>(nullptr, s1b, f2p, fc2b, cur2, 200, 8, 8, 256, 128);

  // layer-2 LIF scan
  k_lif2<<<(64 * 128 + 255) / 256, 256, 0, stream>>>(cur2, beta2, thr2, s2);

  // fco (N=2, VALU dot) + layer-3 LIF scan -> logits
  k_fco<<<(64 * 50 * 2 + 255) / 256, 256, 0, stream>>>(s2, fcow, fcob, cur3);
  k_lif3<<<1, 128, 0, stream>>>(cur3, beta3, thr3, logits);
}
